// Attention_37984690765871
// MI455X (gfx1250) — compile-verified, matrix operand order checked by reading, not run on hardware
//
#include <hip/hip_runtime.h>
#include <hip/hip_bf16.h>

typedef __attribute__((ext_vector_type(16))) _Float16 v16h;
typedef __attribute__((ext_vector_type(8)))  _Float16 v8h;
typedef __attribute__((ext_vector_type(4)))  _Float16 v4h;
typedef __attribute__((ext_vector_type(8)))  float    v8f;

// Padded LDS row stride (in halves): 72*2 = 144 bytes, 16B-aligned rows and
// conflict-free 16B strided reads across 16 lanes (36 dwords mod 64 cycles all banks).
#define LROW 72

// Flash attention: B=4, H=16, L=2048, D=64.
// grid = (L/128, H, B), block = 256 threads = 8 wave32s, each wave owns 16 q-rows.
// K/V tiles are double-buffered in LDS; global loads for tile kb+1 are issued
// before the WMMAs of tile kb so HBM latency hides under matrix-core work.
__global__ __launch_bounds__(256, 1)
void fa_fwd_wmma(const float* __restrict__ Q,
                 const float* __restrict__ K,
                 const float* __restrict__ V,
                 float* __restrict__ O)
{
    constexpr int L = 2048, D = 64, H = 16;
    constexpr int NB = L / 64;               // 32 k-blocks
    constexpr float scale = 0.125f;          // 1/sqrt(64), folded into Q fragments

    const int b    = blockIdx.z;
    const int h    = blockIdx.y;
    const int wave = threadIdx.x >> 5;
    const int lane = threadIdx.x & 31;
    const int t    = threadIdx.x;
    const int q0   = blockIdx.x * 128 + wave * 16;

    const int rm = lane & 15;    // row (A/C) or col (B) index within 16
    const int hi = lane >> 4;    // which K-half this lane holds

    __shared__ _Float16 sK [2][64 * LROW];     // K block, row-major [k][d], f16
    __shared__ _Float16 sVt[2][64 * LROW];     // V block, transposed [d][k], f16
    __shared__ _Float16 sP [8 * 16 * LROW];    // per-wave P staging [16][64], f16

    const size_t bh = (size_t)(b * H + h);
    const float* Qb = Q + bh * (size_t)L * D;
    const float* Kb = K + bh * (size_t)L * D;
    const float* Vb = V + bh * (size_t)L * D;

    // ---- Load Q fragments once (A operand, 16-bit A 16x32 layout), pre-scaled. ----
    // lane holds row rm; halves [0..7] = d(k0+hi*8 .. +7), [8..15] = d(k0+16+hi*8 .. +7)
    v16h aq[2];
    {
        const float* qr = Qb + (size_t)(q0 + rm) * D;
        #pragma unroll
        for (int c = 0; c < 2; ++c) {
            #pragma unroll
            for (int part = 0; part < 2; ++part) {
                const int d0 = c * 32 + part * 16 + hi * 8;
                #pragma unroll
                for (int j = 0; j < 8; ++j)
                    aq[c][part * 8 + j] = (_Float16)(qr[d0 + j] * scale);
            }
        }
    }

    // ---- Flash-attention running state (fp32). ----
    v8f oacc[4];
    #pragma unroll
    for (int i = 0; i < 4; ++i) oacc[i] = (v8f){};
    float m_i[8], l_i[8];
    #pragma unroll
    for (int r = 0; r < 8; ++r) { m_i[r] = -1e30f; l_i[r] = 0.f; }

    _Float16* pbase = &sP[wave * 16 * LROW];

    // Per-thread staging registers for the pipelined K/V tile (4 float4 each).
    float4 kreg[4], vreg[4];
    const int srow[4] = { (t + 0*256) >> 4, (t + 1*256) >> 4,
                          (t + 2*256) >> 4, (t + 3*256) >> 4 };
    const int sc4 = t & 15;   // same 16-float4-per-row phase for all i

    // ---- Prologue: fetch tile 0 into registers, stage into buffer 0. ----
    {
        const float* Kt = Kb;
        const float* Vt = Vb;
        #pragma unroll
        for (int i = 0; i < 4; ++i) {
            kreg[i] = *(const float4*)(Kt + srow[i] * D + sc4 * 4);
            vreg[i] = *(const float4*)(Vt + srow[i] * D + sc4 * 4);
        }
        #pragma unroll
        for (int i = 0; i < 4; ++i) {
            v4h kh = { (_Float16)kreg[i].x, (_Float16)kreg[i].y,
                       (_Float16)kreg[i].z, (_Float16)kreg[i].w };
            *(v4h*)&sK[0][srow[i] * LROW + sc4 * 4] = kh;
            sVt[0][(sc4 * 4 + 0) * LROW + srow[i]] = (_Float16)vreg[i].x;
            sVt[0][(sc4 * 4 + 1) * LROW + srow[i]] = (_Float16)vreg[i].y;
            sVt[0][(sc4 * 4 + 2) * LROW + srow[i]] = (_Float16)vreg[i].z;
            sVt[0][(sc4 * 4 + 3) * LROW + srow[i]] = (_Float16)vreg[i].w;
        }
    }
    __syncthreads();

    for (int kb = 0; kb < NB; ++kb) {
        const int cur = kb & 1;
        const int nxt = cur ^ 1;

        // ---- Issue next tile's global loads NOW; consumed after compute. ----
        if (kb + 1 < NB) {
            const float* Kt = Kb + (size_t)(kb + 1) * 64 * D;
            const float* Vt = Vb + (size_t)(kb + 1) * 64 * D;
            #pragma unroll
            for (int i = 0; i < 4; ++i) {
                kreg[i] = *(const float4*)(Kt + srow[i] * D + sc4 * 4);
                vreg[i] = *(const float4*)(Vt + srow[i] * D + sc4 * 4);
            }
            if (kb + 2 < NB) {    // keep L2 warm one more tile ahead
                __builtin_prefetch(Kt + 64 * D + t * 16, 0, 3);
                __builtin_prefetch(Vt + 64 * D + t * 16, 0, 3);
            }
        }

        // ---- S = (Q*scale) * K^T : 16 x 64 tile, 8 WMMAs. ----
        v8f sacc[4];
        #pragma unroll
        for (int n = 0; n < 4; ++n) {
            v8f c = (v8f){};
            #pragma unroll
            for (int cc = 0; cc < 2; ++cc) {
                v16h bk;                          // B operand: lane holds k-row n*16+rm
                const int nrow = n * 16 + rm;
                #pragma unroll
                for (int part = 0; part < 2; ++part) {
                    v8h x = *(const v8h*)&sK[cur][nrow * LROW + cc * 32 + part * 16 + hi * 8];
                    #pragma unroll
                    for (int j = 0; j < 8; ++j) bk[part * 8 + j] = x[j];
                }
                c = __builtin_amdgcn_wmma_f32_16x16x32_f16(
                        false, aq[cc], false, bk, (short)0, c, false, false);
            }
            sacc[n] = c;
        }

        // ---- Online softmax (C layout: VGPR r -> row r+hi*8, lane -> col). ----
        float mnew[8], sf[8], rs[8];
        #pragma unroll
        for (int r = 0; r < 8; ++r) {
            float mx = sacc[0][r];
            #pragma unroll
            for (int n = 1; n < 4; ++n) mx = fmaxf(mx, sacc[n][r]);
            #pragma unroll
            for (int off = 1; off < 16; off <<= 1)
                mx = fmaxf(mx, __shfl_xor(mx, off, 32));
            mnew[r] = fmaxf(m_i[r], mx);
            sf[r]   = __expf(m_i[r] - mnew[r]);
            m_i[r]  = mnew[r];
            rs[r]   = 0.f;
        }
        #pragma unroll
        for (int n = 0; n < 4; ++n) {
            #pragma unroll
            for (int r = 0; r < 8; ++r) {
                float p = __expf(sacc[n][r] - mnew[r]);
                rs[r] += p;
                // wave-private staging of P in row-major f16 for A-layout reload
                pbase[(r + hi * 8) * LROW + n * 16 + rm] = (_Float16)p;
            }
        }
        #pragma unroll
        for (int r = 0; r < 8; ++r) {
            float s = rs[r];
            #pragma unroll
            for (int off = 1; off < 16; off <<= 1)
                s += __shfl_xor(s, off, 32);
            l_i[r] = l_i[r] * sf[r] + s;
        }
        #pragma unroll
        for (int dt = 0; dt < 4; ++dt)
            #pragma unroll
            for (int r = 0; r < 8; ++r) oacc[dt][r] *= sf[r];

        // ---- O += P * V : 8 WMMAs (same-wave LDS RAW -> in-order DS, no barrier). ----
        #pragma unroll
        for (int dt = 0; dt < 4; ++dt) {
            v8f c = oacc[dt];
            #pragma unroll
            for (int cc = 0; cc < 2; ++cc) {
                v16h ap, bv;
                #pragma unroll
                for (int part = 0; part < 2; ++part) {
                    const int k0 = cc * 32 + part * 16 + hi * 8;
                    v8h x = *(const v8h*)&pbase[rm * LROW + k0];
                    v8h y = *(const v8h*)&sVt[cur][(dt * 16 + rm) * LROW + k0];
                    #pragma unroll
                    for (int j = 0; j < 8; ++j) {
                        ap[part * 8 + j] = x[j];
                        bv[part * 8 + j] = y[j];
                    }
                }
                c = __builtin_amdgcn_wmma_f32_16x16x32_f16(
                        false, ap, false, bv, (short)0, c, false, false);
            }
            oacc[dt] = c;
        }

        // ---- Stage next tile into the other buffer (safe: last read of `nxt`
        //      finished before the barrier at the end of iteration kb-1). ----
        if (kb + 1 < NB) {
            #pragma unroll
            for (int i = 0; i < 4; ++i) {
                v4h kh = { (_Float16)kreg[i].x, (_Float16)kreg[i].y,
                           (_Float16)kreg[i].z, (_Float16)kreg[i].w };
                *(v4h*)&sK[nxt][srow[i] * LROW + sc4 * 4] = kh;
                sVt[nxt][(sc4 * 4 + 0) * LROW + srow[i]] = (_Float16)vreg[i].x;
                sVt[nxt][(sc4 * 4 + 1) * LROW + srow[i]] = (_Float16)vreg[i].y;
                sVt[nxt][(sc4 * 4 + 2) * LROW + srow[i]] = (_Float16)vreg[i].z;
                sVt[nxt][(sc4 * 4 + 3) * LROW + srow[i]] = (_Float16)vreg[i].w;
            }
        }
        __syncthreads();   // next-tile stores visible; reads of `cur` complete
    }

    // ---- Epilogue: normalize and write out[b][q][h*64 + d]. ----
    float* Ob = O + (size_t)b * L * (H * D) + (size_t)h * D;
    #pragma unroll
    for (int r = 0; r < 8; ++r) {
        const int   q     = q0 + r + hi * 8;
        const float inv_l = 1.0f / l_i[r];
        #pragma unroll
        for (int dt = 0; dt < 4; ++dt)
            Ob[(size_t)q * (H * D) + dt * 16 + rm] = oacc[dt][r] * inv_l;
    }
}

extern "C" void kernel_launch(void* const* d_in, const int* in_sizes, int n_in,
                              void* d_out, int out_size, void* d_ws, size_t ws_size,
                              hipStream_t stream) {
    const float* Q = (const float*)d_in[0];
    const float* K = (const float*)d_in[1];
    const float* V = (const float*)d_in[2];
    float*       O = (float*)d_out;
    // B=4, H=16, L=2048, D=64; 128 q-rows per workgroup.
    dim3 grid(2048 / 128, 16, 4);
    dim3 block(256);
    fa_fwd_wmma<<<grid, block, 0, stream>>>(Q, K, V, O);
}
